// TEM_4535485464580
// MI455X (gfx1250) — compile-verified, matrix operand order checked by reading, not run on hardware
//
#include <hip/hip_runtime.h>
#include <hip/hip_bf16.h>
#include <math.h>

// ---------------- model constants ----------------
#define N_FREQ   5
#define N_PHASE  8
#define S_COMP   10
#define G_PER    24
#define S_SIZE   45
#define N_ACT    4
#define G_DIM    120            // N_FREQ*G_PER
#define P_DIM    400            // N_FREQ*N_PHASE*S_COMP
#define B_DIM    64
#define SEQ_LEN  16
#define KAPPA    0.8f
#define FORGET_L 0.98990100f    // 0.99*0.9999
#define ETA      0.5f
#define T_HID    100
#define GG       14400          // G*G
#define NTILES_N 900            // GG/16
#define NTILES_M 4              // B/16
#define KCHUNKS  4              // ceil(100/32)

typedef __attribute__((ext_vector_type(16))) _Float16 v16h;
typedef __attribute__((ext_vector_type(8)))  float    v8f;

__device__ __forceinline__ float f_act(float z) {
    float y = (z >= 0.f) ? z : 0.01f * z;          // leaky_relu(0.01)
    return fminf(1.f, fmaxf(-1.f, y));             // clip [-1,1]
}
__device__ __forceinline__ float elu_f(float x) {
    return (x > 0.f) ? x : expm1f(x);
}
// K index inside a 32-wide chunk for fp16 A/B fragments (ISA 7.12.2, wave32)
__device__ __forceinline__ int kmap16(int e, int half) {
    int v = e >> 1;
    int kb = (v < 4) ? (half * 8 + 2 * v) : (16 + half * 8 + 2 * (v - 4));
    return kb + (e & 1);
}

// ---------------- kernels ----------------

// h[b,k] = tanh(sum_a d_t[b,a]*t_W1[a,k])   (64x100)
__global__ void k_tanh_dW1(const float* __restrict__ d_t, const float* __restrict__ tW1,
                           float* __restrict__ h) {
    int idx = blockIdx.x * blockDim.x + threadIdx.x;
    if (idx >= B_DIM * T_HID) return;
    int b = idx / T_HID, k = idx % T_HID;
    float acc = 0.f;
    #pragma unroll
    for (int a = 0; a < N_ACT; ++a) acc += d_t[b * N_ACT + a] * tW1[a * T_HID + k];
    h[idx] = tanhf(acc);
}

// pack h[64,100] -> fp16 A-fragments: apack[((mi*4+kc)*32+lane)*16+e]
__global__ void k_apack(const float* __restrict__ h, _Float16* __restrict__ apack) {
    int idx = blockIdx.x * blockDim.x + threadIdx.x;          // 4*4*32*16 = 8192
    if (idx >= NTILES_M * KCHUNKS * 32 * 16) return;
    int e    = idx & 15;
    int lane = (idx >> 4) & 31;
    int kc   = (idx >> 9) & 3;
    int mi   = idx >> 11;
    int half = lane >> 4, rc = lane & 15;
    int k = kc * 32 + kmap16(e, half);
    apack[idx] = (k < T_HID) ? (_Float16)h[(mi * 16 + rc) * T_HID + k] : (_Float16)0.f;
}

// pack t_W2[100,14400] -> fp16 B-fragments (ONCE per launch, reused all 16 steps):
// bpack[((kc*900+ni)*32+lane)*16+e]
__global__ void k_bpack(const float* __restrict__ tW2, _Float16* __restrict__ bpack) {
    int idx = blockIdx.x * blockDim.x + threadIdx.x;          // 4*900*32*16 = 1,843,200
    if (idx >= KCHUNKS * NTILES_N * 32 * 16) return;
    int e    = idx & 15;
    int lane = (idx >> 4) & 31;
    int rest = idx >> 9;
    int ni   = rest % NTILES_N;
    int kc   = rest / NTILES_N;
    int half = lane >> 4, col = lane & 15;
    int k = kc * 32 + kmap16(e, half);
    bpack[idx] = (k < T_HID) ? (_Float16)tW2[(size_t)k * GG + ni * 16 + col] : (_Float16)0.f;
}

// t_mat[64,14400] = h @ t_W2 via WMMA on pre-packed fragments.
// One 16x16 tile per wave32; inner loop = 2x b128 fragment loads + v_wmma.
__global__ void k_tmat_wmma(const _Float16* __restrict__ apack,
                            const _Float16* __restrict__ bpack,
                            float* __restrict__ C) {
    int wave = (blockIdx.x * blockDim.x + threadIdx.x) >> 5;
    int lane = threadIdx.x & 31;
    int mi = wave / NTILES_N;                  // 0..3
    int ni = wave % NTILES_N;
    if (mi >= NTILES_M) return;
    int half = lane >> 4, rc = lane & 15;
    const v16h* ap = reinterpret_cast<const v16h*>(apack);
    const v16h* bp = reinterpret_cast<const v16h*>(bpack);
    v8f acc = {};
    #pragma unroll
    for (int kc = 0; kc < KCHUNKS; ++kc) {
        v16h af = ap[(mi * KCHUNKS + kc) * 32 + lane];
        v16h bf = bp[(kc * NTILES_N + ni) * 32 + lane];
        acc = __builtin_amdgcn_wmma_f32_16x16x32_f16(false, af, false, bf,
                                                     (short)0, acc, false, false);
    }
    #pragma unroll
    for (int r = 0; r < 8; ++r)
        C[(size_t)(mi * 16 + r + 8 * half) * GG + ni * 16 + rc] = acc[r];
}

// g_gen[b,i] = f_act(g[b,i] + sum_j t_mat[b,i,j]*g[b,j])
__global__ void k_gen_g(const float* __restrict__ g, const float* __restrict__ tmat,
                        float* __restrict__ ggen) {
    int idx = blockIdx.x * blockDim.x + threadIdx.x;
    if (idx >= B_DIM * G_DIM) return;
    int b = idx / G_DIM, i = idx % G_DIM;
    const float* T  = tmat + (size_t)b * GG + (size_t)i * G_DIM;
    const float* gb = g + b * G_DIM;
    float acc = gb[i];
    for (int j = 0; j < G_DIM; ++j) acc += T[j] * gb[j];
    ggen[idx] = f_act(acc);
}

// per-frequency MLP layer1: out[b,f,h]=elu(sum_i in*W1[f,i,h]+b1[f,h]); H=48
__global__ void k_pf_l1(const float* __restrict__ in, const float* __restrict__ W1,
                        const float* __restrict__ b1, float* __restrict__ out,
                        int IN, int perF) {
    int idx = blockIdx.x * blockDim.x + threadIdx.x;
    if (idx >= B_DIM * N_FREQ * 48) return;
    int b = idx / (N_FREQ * 48);
    int r = idx % (N_FREQ * 48);
    int f = r / 48, hI = r % 48;
    float acc = b1[f * 48 + hI];
    for (int i = 0; i < IN; ++i) {
        float xin = perF ? in[b * N_FREQ * IN + f * IN + i] : in[b * IN + i];
        acc += xin * W1[(f * IN + i) * 48 + hI];
    }
    out[idx] = elu_f(acc);
}

// per-frequency MLP layer2: out[b, f*24+o]; mode 0: exp(tanh(z)); mode 1: f_act(z)
__global__ void k_pf_l2(const float* __restrict__ hh, const float* __restrict__ W2,
                        const float* __restrict__ b2, float* __restrict__ out, int mode) {
    int idx = blockIdx.x * blockDim.x + threadIdx.x;
    if (idx >= B_DIM * N_FREQ * G_PER) return;
    int b = idx / (N_FREQ * G_PER);
    int r = idx % (N_FREQ * G_PER);
    int f = r / G_PER, o = r % G_PER;
    float acc = b2[f * G_PER + o];
    for (int h = 0; h < 48; ++h)
        acc += hh[b * N_FREQ * 48 + f * 48 + h] * W2[(f * 48 + h) * G_PER + o];
    out[b * G_DIM + f * G_PER + o] = mode ? f_act(acc) : expf(tanhf(acc));
}

// dense: out[b,n] = act(sum_k in[b,k]*W[k,n]+bias[n]); act 0=none,1=elu
__global__ void k_dense(const float* __restrict__ in, const float* __restrict__ W,
                        const float* __restrict__ bias, float* __restrict__ out,
                        int K, int N, int act) {
    int idx = blockIdx.x * blockDim.x + threadIdx.x;
    if (idx >= B_DIM * N) return;
    int b = idx / N, n = idx % N;
    float acc = bias[n];
    for (int k = 0; k < K; ++k) acc += in[b * K + k] * W[k * N + n];
    out[idx] = act ? elu_f(acc) : acc;
}

// x_filt[f,b,s] = a*x_filt + (1-a)*xc[b,s],  a = sigmoid(gamma[f])
__global__ void k_filt(float* __restrict__ xf, const float* __restrict__ xc,
                       const float* __restrict__ gamma) {
    int idx = blockIdx.x * blockDim.x + threadIdx.x;
    if (idx >= N_FREQ * B_DIM * S_COMP) return;
    int f = idx / (B_DIM * S_COMP);
    int rem = idx % (B_DIM * S_COMP);
    int b = rem / S_COMP, s = rem % S_COMP;
    float a = 1.f / (1.f + expf(-gamma[f]));
    xf[idx] = a * xf[idx] + (1.f - a) * xc[b * S_COMP + s];
}

// x2p[b, (f*8+ph)*10+s] = w_p[f]*x_filt[f,b,s]
__global__ void k_x2p(const float* __restrict__ xf, const float* __restrict__ w_p,
                      float* __restrict__ x2p) {
    int idx = blockIdx.x * blockDim.x + threadIdx.x;
    if (idx >= B_DIM * P_DIM) return;
    int b = idx / P_DIM, i = idx % P_DIM;
    int f = i / (N_PHASE * S_COMP);
    int s = i % S_COMP;
    x2p[idx] = w_p[f] * xf[f * B_DIM * S_COMP + b * S_COMP + s];
}

// one attractor iteration: pout = f_act(inp + 0.8*M@pin); first -> pin := f_act(inp)
// Wave-per-row, lane-parallel dot, p staged in LDS; prefetch next row (global_prefetch_b8).
__global__ void k_attr_iter(const float* __restrict__ inp, const float* __restrict__ pin,
                            const float* __restrict__ M, float* __restrict__ pout, int first) {
    __shared__ float ps[P_DIM];
    int b = blockIdx.x;
    int tid = threadIdx.x;                         // 256 threads = 8 waves
    for (int j = tid; j < P_DIM; j += 256) {
        float v = pin[b * P_DIM + j];
        if (first) v = f_act(inp[b * P_DIM + j]);
        ps[j] = v;
    }
    __syncthreads();
    int wave = tid >> 5, lane = tid & 31;
    const float* Mb = M + (size_t)b * P_DIM * P_DIM;
    for (int r = wave; r < P_DIM; r += 8) {
        if (r + 8 < P_DIM)   // prime L2->WGP path one row ahead
            __builtin_prefetch(&Mb[(size_t)(r + 8) * P_DIM + lane * 4], 0, 1);
        float acc = 0.f;
        for (int j = lane; j < P_DIM; j += 32)
            acc += Mb[(size_t)r * P_DIM + j] * ps[j];
        #pragma unroll
        for (int off = 16; off > 0; off >>= 1)
            acc += __shfl_xor(acc, off, 32);
        if (lane == 0)
            pout[b * P_DIM + r] = f_act(inp[b * P_DIM + r] + KAPPA * acc);
    }
}

// sens[b,f,ph] = mean_s p_x[b,(f*8+ph)*10+s]
__global__ void k_sens(const float* __restrict__ px, float* __restrict__ sens) {
    int idx = blockIdx.x * blockDim.x + threadIdx.x;
    if (idx >= B_DIM * N_FREQ * N_PHASE) return;
    int b = idx / (N_FREQ * N_PHASE), fp = idx % (N_FREQ * N_PHASE);
    float acc = 0.f;
    #pragma unroll
    for (int s = 0; s < S_COMP; ++s) acc += px[b * P_DIM + fp * S_COMP + s];
    sens[idx] = acc * 0.1f;
}

// pooled2[b,s] = w_x0 * mean_fp p[b, fp*10+s] + b_x[s]
__global__ void k_fxpool(const float* __restrict__ p, const float* __restrict__ w_x,
                         const float* __restrict__ b_x, float* __restrict__ out) {
    int idx = blockIdx.x * blockDim.x + threadIdx.x;
    if (idx >= B_DIM * S_COMP) return;
    int b = idx / S_COMP, s = idx % S_COMP;
    float acc = 0.f;
    for (int fp = 0; fp < N_FREQ * N_PHASE; ++fp) acc += p[b * P_DIM + fp * S_COMP + s];
    out[idx] = w_x[0] * acc * (1.f / (N_FREQ * N_PHASE)) + b_x[s];
}

// softmax over logits[b,:45], err = sum((x_t - softmax)^2) -> e2[b]={err,1}
__global__ void k_softmax_err(const float* __restrict__ logits, const float* __restrict__ xt,
                              float* __restrict__ e2) {
    int b = blockIdx.x;
    if (threadIdx.x != 0) return;
    const float* L = logits + b * S_SIZE;
    const float* X = xt + b * S_SIZE;
    float mx = L[0];
    for (int s = 1; s < S_SIZE; ++s) mx = fmaxf(mx, L[s]);
    float ex[S_SIZE]; float sum = 0.f;
    for (int s = 0; s < S_SIZE; ++s) { ex[s] = expf(L[s] - mx); sum += ex[s]; }
    float inv = 1.f / sum, err = 0.f;
    for (int s = 0; s < S_SIZE; ++s) { float d = X[s] - ex[s] * inv; err += d * d; }
    e2[b * 2] = err; e2[b * 2 + 1] = 1.f;
}

// precision-weighted combination
__global__ void k_combine(const float* __restrict__ ggen, const float* __restrict__ sgen,
                          const float* __restrict__ mu, const float* __restrict__ sp2g,
                          float* __restrict__ gnew) {
    int idx = blockIdx.x * blockDim.x + threadIdx.x;
    if (idx >= B_DIM * G_DIM) return;
    float pr1 = 1.f / (sgen[idx] * sgen[idx] + 1e-6f);
    float pr2 = 1.f / (sp2g[idx] * sp2g[idx] + 1e-6f);
    gnew[idx] = (ggen[idx] * pr1 + mu[idx] * pr2) / (pr1 + pr2);
}

// g2p: out[b,(f*8+ph)*10+s] = sum_g gin[b,f*24+g]*W[f,g,ph]
__global__ void k_g2p(const float* __restrict__ gin, const float* __restrict__ W,
                      float* __restrict__ out) {
    int idx = blockIdx.x * blockDim.x + threadIdx.x;
    if (idx >= B_DIM * P_DIM) return;
    int b = idx / P_DIM, i = idx % P_DIM;
    int f = i / (N_PHASE * S_COMP);
    int ph = (i % (N_PHASE * S_COMP)) / S_COMP;
    float acc = 0.f;
    #pragma unroll
    for (int g = 0; g < G_PER; ++g)
        acc += gin[b * G_DIM + f * G_PER + g] * W[(f * G_PER + g) * N_PHASE + ph];
    out[idx] = acc;
}

// p = f_act(g2p_inf * x2p)
__global__ void k_pmul(const float* __restrict__ a, const float* __restrict__ bb,
                       float* __restrict__ out) {
    int idx = blockIdx.x * blockDim.x + threadIdx.x;
    if (idx >= B_DIM * P_DIM) return;
    out[idx] = f_act(a[idx] * bb[idx]);
}

// M = lambda*M + eta*(p-q)[r]*(p+q)[c] (* mask)
__global__ void k_hebb(float* __restrict__ M, const float* __restrict__ p,
                       const float* __restrict__ q, int useMask) {
    size_t idx = (size_t)blockIdx.x * blockDim.x + threadIdx.x;
    if (idx >= (size_t)B_DIM * P_DIM * P_DIM) return;
    int b = idx / (P_DIM * P_DIM);
    int rem = idx % (P_DIM * P_DIM);
    int r = rem / P_DIM, c = rem % P_DIM;
    float w = (p[b * P_DIM + r] - q[b * P_DIM + r]) * (p[b * P_DIM + c] + q[b * P_DIM + c]);
    if (useMask && (r / (N_PHASE * S_COMP) < c / (N_PHASE * S_COMP))) w = 0.f;
    M[idx] = FORGET_L * M[idx] + ETA * w;
}

__global__ void k_clip(float* __restrict__ M, size_t n) {
    size_t idx = (size_t)blockIdx.x * blockDim.x + threadIdx.x;
    if (idx >= n) return;
    M[idx] = fminf(1.f, fmaxf(-1.f, M[idx]));
}

// ---------------- host driver ----------------
#define GRID(n, b) ((int)(((n) + (b) - 1) / (b)))

extern "C" void kernel_launch(void* const* d_in, const int* in_sizes, int n_in,
                              void* d_out, int out_size, void* d_ws, size_t ws_size,
                              hipStream_t stream) {
    const float* x      = (const float*)d_in[0];
    const float* dd     = (const float*)d_in[1];
    const float* g0     = (const float*)d_in[2];
    const float* xf0    = (const float*)d_in[3];
    const float* hg0    = (const float*)d_in[4];
    const float* hi0    = (const float*)d_in[5];
    const float* gamma  = (const float*)d_in[6];
    const float* w_x    = (const float*)d_in[7];
    const float* b_x    = (const float*)d_in[8];
    const float* w_p    = (const float*)d_in[9];
    const float* t_W1   = (const float*)d_in[10];
    const float* t_W2   = (const float*)d_in[11];
    const float* c_W1   = (const float*)d_in[12];
    const float* c_b1   = (const float*)d_in[13];
    const float* c_W2   = (const float*)d_in[14];
    const float* c_b2   = (const float*)d_in[15];
    const float* cs_W1  = (const float*)d_in[16];
    const float* cs_b1  = (const float*)d_in[17];
    const float* cs_W2  = (const float*)d_in[18];
    const float* cs_b2  = (const float*)d_in[19];
    const float* mu_W1  = (const float*)d_in[20];
    const float* mu_b1  = (const float*)d_in[21];
    const float* mu_W2  = (const float*)d_in[22];
    const float* mu_b2  = (const float*)d_in[23];
    const float* ls_W1  = (const float*)d_in[24];
    const float* ls_b1  = (const float*)d_in[25];
    const float* ls_W2  = (const float*)d_in[26];
    const float* ls_b2  = (const float*)d_in[27];
    const float* gg_W1  = (const float*)d_in[28];
    const float* gg_b1  = (const float*)d_in[29];
    const float* gg_W2  = (const float*)d_in[30];
    const float* gg_b2  = (const float*)d_in[31];
    const float* W_g2p  = (const float*)d_in[32];

    float* out = (float*)d_out;
    const size_t LOGITS_SZ = (size_t)3 * SEQ_LEN * B_DIM * S_SIZE;   // 138240
    const size_t G_SZ      = (size_t)B_DIM * G_DIM;                  // 7680
    const size_t M_SZ      = (size_t)B_DIM * P_DIM * P_DIM;          // 10,240,000
    float* logits_out = out;
    float* gfin_out   = out + LOGITS_SZ;
    float* Mgen       = out + LOGITS_SZ + G_SZ;          // M_gen lives in output (L2-resident)
    float* Minf       = Mgen + M_SZ;

    // workspace layout: fp16 fragment packs first (32B-aligned at ws base), floats after
    const size_t BPACK_H = (size_t)KCHUNKS * NTILES_N * 32 * 16;     // 1,843,200 halves
    const size_t APACK_H = (size_t)NTILES_M * KCHUNKS * 32 * 16;     // 8,192 halves
    _Float16* bpack = (_Float16*)d_ws;
    _Float16* apack = bpack + BPACK_H;
    float* ws = (float*)(apack + APACK_H);

    float* gA      = ws;            size_t o = G_SZ;
    float* gB      = ws + o;        o += G_SZ;
    float* x_filt  = ws + o;        o += N_FREQ * B_DIM * S_COMP;
    float* h_t     = ws + o;        o += B_DIM * T_HID;
    float* t_mat   = ws + o;        o += (size_t)B_DIM * GG;
    float* g_gen   = ws + o;        o += G_SZ;
    float* s_gen   = ws + o;        o += G_SZ;
    float* mu_p2g  = ws + o;        o += G_SZ;
    float* s_p2g   = ws + o;        o += G_SZ;
    float* xh      = ws + o;        o += B_DIM * 200;
    float* xc      = ws + o;        o += B_DIM * S_COMP;
    float* x2p     = ws + o;        o += B_DIM * P_DIM;
    float* p_x     = ws + o;        o += B_DIM * P_DIM;
    float* p_v     = ws + o;        o += B_DIM * P_DIM;
    float* p_g     = ws + o;        o += B_DIM * P_DIM;
    float* p_gt    = ws + o;        o += B_DIM * P_DIM;
    float* g2pi    = ws + o;        o += B_DIM * P_DIM;
    float* g2pg    = ws + o;        o += B_DIM * P_DIM;
    float* tmpP    = ws + o;        o += B_DIM * P_DIM;
    float* sens    = ws + o;        o += B_DIM * N_FREQ * N_PHASE;
    float* hh      = ws + o;        o += B_DIM * N_FREQ * 48;
    float* pooled2 = ws + o;        o += B_DIM * S_COMP;
    float* fxh     = ws + o;        o += B_DIM * 200;
    float* fxlog   = ws + o;        o += B_DIM * S_SIZE;
    float* e2      = ws + o;        o += B_DIM * 2;

    // init state (re-done every call -> deterministic, graph-capture safe)
    hipMemcpyAsync(Mgen, hg0, M_SZ * sizeof(float), hipMemcpyDeviceToDevice, stream);
    hipMemcpyAsync(Minf, hi0, M_SZ * sizeof(float), hipMemcpyDeviceToDevice, stream);
    hipMemcpyAsync(gA, g0, G_SZ * sizeof(float), hipMemcpyDeviceToDevice, stream);
    hipMemcpyAsync(x_filt, xf0, N_FREQ * B_DIM * S_COMP * sizeof(float),
                   hipMemcpyDeviceToDevice, stream);

    // pack t_W2 into fp16 B-fragments ONCE; amortized over all 16 steps
    k_bpack<<<GRID(BPACK_H, 256), 256, 0, stream>>>(t_W2, bpack);

    float* gcur = gA;
    float* gnext = gB;

    for (int t = 0; t < SEQ_LEN; ++t) {
        const float* x_t = x + (size_t)t * B_DIM * S_SIZE;
        const float* d_t = dd + (size_t)t * B_DIM * N_ACT;

        // generative grid transition
        k_tanh_dW1<<<GRID(B_DIM * T_HID, 256), 256, 0, stream>>>(d_t, t_W1, h_t);
        k_apack<<<GRID(APACK_H, 256), 256, 0, stream>>>(h_t, apack);
        k_tmat_wmma<<<450, 256, 0, stream>>>(apack, bpack, t_mat);   // 3600 waves, WMMA
        k_gen_g<<<GRID(B_DIM * G_DIM, 256), 256, 0, stream>>>(gcur, t_mat, g_gen);
        k_pf_l1<<<GRID(B_DIM * N_FREQ * 48, 256), 256, 0, stream>>>(g_gen, gg_W1, gg_b1, hh, G_PER, 1);
        k_pf_l2<<<GRID(B_DIM * G_DIM, 256), 256, 0, stream>>>(hh, gg_W2, gg_b2, s_gen, 0);

        // sensory compression + temporal filtering
        k_dense<<<GRID(B_DIM * 200, 256), 256, 0, stream>>>(x_t, c_W1, c_b1, xh, S_SIZE, 200, 1);
        k_dense<<<GRID(B_DIM * S_COMP, 256), 256, 0, stream>>>(xh, c_W2, c_b2, xc, 200, S_COMP, 0);
        k_filt<<<GRID(N_FREQ * B_DIM * S_COMP, 256), 256, 0, stream>>>(x_filt, xc, gamma);
        k_x2p<<<GRID(B_DIM * P_DIM, 256), 256, 0, stream>>>(x_filt, w_p, x2p);

        // p2g: pattern completion from inverse memory
        k_attr_iter<<<B_DIM, 256, 0, stream>>>(x2p, x2p, Minf, tmpP, 1);
        k_attr_iter<<<B_DIM, 256, 0, stream>>>(x2p, tmpP, Minf, p_x, 0);
        k_sens<<<GRID(B_DIM * N_FREQ * N_PHASE, 256), 256, 0, stream>>>(p_x, sens);
        k_pf_l1<<<GRID(B_DIM * N_FREQ * 48, 256), 256, 0, stream>>>(sens, mu_W1, mu_b1, hh, N_PHASE, 1);
        k_pf_l2<<<GRID(B_DIM * G_DIM, 256), 256, 0, stream>>>(hh, mu_W2, mu_b2, mu_p2g, 1);

        // fx(p_x) -> softmax error -> sigma_p2g
        k_fxpool<<<GRID(B_DIM * S_COMP, 256), 256, 0, stream>>>(p_x, w_x, b_x, pooled2);
        k_dense<<<GRID(B_DIM * 200, 256), 256, 0, stream>>>(pooled2, cs_W1, cs_b1, fxh, S_COMP, 200, 1);
        k_dense<<<GRID(B_DIM * S_SIZE, 256), 256, 0, stream>>>(fxh, cs_W2, cs_b2, fxlog, 200, S_SIZE, 0);
        k_softmax_err<<<B_DIM, 32, 0, stream>>>(fxlog, x_t, e2);
        k_pf_l1<<<GRID(B_DIM * N_FREQ * 48, 256), 256, 0, stream>>>(e2, ls_W1, ls_b1, hh, 2, 0);
        k_pf_l2<<<GRID(B_DIM * G_DIM, 256), 256, 0, stream>>>(hh, ls_W2, ls_b2, s_p2g, 0);

        // precision-weighted fusion
        k_combine<<<GRID(B_DIM * G_DIM, 256), 256, 0, stream>>>(g_gen, s_gen, mu_p2g, s_p2g, gnext);

        // infer_p / generation
        k_g2p<<<GRID(B_DIM * P_DIM, 256), 256, 0, stream>>>(gnext, W_g2p, g2pi);
        k_g2p<<<GRID(B_DIM * P_DIM, 256), 256, 0, stream>>>(g_gen, W_g2p, g2pg);
        k_pmul<<<GRID(B_DIM * P_DIM, 256), 256, 0, stream>>>(g2pi, x2p, p_v);
        k_attr_iter<<<B_DIM, 256, 0, stream>>>(g2pi, g2pi, Mgen, tmpP, 1);
        k_attr_iter<<<B_DIM, 256, 0, stream>>>(g2pi, tmpP, Mgen, p_g, 0);
        k_attr_iter<<<B_DIM, 256, 0, stream>>>(g2pg, g2pg, Mgen, tmpP, 1);
        k_attr_iter<<<B_DIM, 256, 0, stream>>>(g2pg, tmpP, Mgen, p_gt, 0);

        // logits_p / logits_g / logits_gt straight into d_out[k,t,:,:]
        const float* ps[3] = { p_v, p_g, p_gt };
        for (int kk = 0; kk < 3; ++kk) {
            float* dst = logits_out + ((size_t)kk * SEQ_LEN + t) * B_DIM * S_SIZE;
            k_fxpool<<<GRID(B_DIM * S_COMP, 256), 256, 0, stream>>>(ps[kk], w_x, b_x, pooled2);
            k_dense<<<GRID(B_DIM * 200, 256), 256, 0, stream>>>(pooled2, cs_W1, cs_b1, fxh, S_COMP, 200, 1);
            k_dense<<<GRID(B_DIM * S_SIZE, 256), 256, 0, stream>>>(fxh, cs_W2, cs_b2, dst, 200, S_SIZE, 0);
        }

        // Hebbian fast-weight writes (in place in d_out region; L2 resident)
        k_hebb<<<GRID(M_SZ, 256), 256, 0, stream>>>(Mgen, p_v, p_g, 1);
        k_hebb<<<GRID(M_SZ, 256), 256, 0, stream>>>(Minf, p_v, p_x, 0);

        float* tsw = gcur; gcur = gnext; gnext = tsw;
    }

    hipMemcpyAsync(gfin_out, gcur, G_SZ * sizeof(float), hipMemcpyDeviceToDevice, stream);
    k_clip<<<GRID(M_SZ, 256), 256, 0, stream>>>(Mgen, M_SZ);
    k_clip<<<GRID(M_SZ, 256), 256, 0, stream>>>(Minf, M_SZ);

    (void)in_sizes; (void)n_in; (void)out_size; (void)ws_size;
}
